// FastHoloBlock_85323820303029
// MI455X (gfx1250) — matrix operand order repaired and checked by reading, not compiled, not run on hardware
//
#include <hip/hip_runtime.h>
#include <hip/hip_bf16.h>

typedef unsigned short ushort_t;
typedef unsigned int   uint_t;

typedef __bf16 v16bf __attribute__((ext_vector_type(16)));
typedef float  v8f   __attribute__((ext_vector_type(8)));
typedef float  v2f   __attribute__((ext_vector_type(2)));
typedef int    v4i   __attribute__((ext_vector_type(4)));
typedef int    v8i   __attribute__((ext_vector_type(8)));

struct alignas(16) Q4 { uint_t x, y, z, w; };
struct Q8 { Q4 a, b; };

typedef __attribute__((address_space(3))) char lds_char;

#define EDIM 1024
#define MDIM 256
#define BB   4
#define SS   2048
#define ROWS_TOT (BB*SS)   // 8192
#define FFDIM (4*EDIM)     // 4096

// ---------------------------------------------------------------- helpers
__device__ __forceinline__ ushort_t f2bf(float f) {
  uint_t u = __builtin_bit_cast(uint_t, f);
  u += 0x7FFFu + ((u >> 16) & 1u);          // round-to-nearest-even
  return (ushort_t)(u >> 16);
}
__device__ __forceinline__ float bf2f(ushort_t h) {
  return __builtin_bit_cast(float, (uint_t)h << 16);
}
__device__ __forceinline__ float sigmoidf_(float x) { return 1.0f / (1.0f + __expf(-x)); }
__device__ __forceinline__ v16bf to_v16bf(Q4 a, Q4 b) { Q8 q{a, b}; return __builtin_bit_cast(v16bf, q); }

// LDS byte address (wave-relative) of a generic pointer known to be in LDS
__device__ __forceinline__ unsigned lds_addr32(const void* p) {
  return (unsigned)(unsigned long long)(const lds_char*)p;
}
// CDNA5 async global->LDS copies (ASYNCcnt)
__device__ __forceinline__ void async_b128(unsigned lds, const void* g) {
  asm volatile("global_load_async_to_lds_b128 %0, %1, off" :: "v"(lds), "v"(g) : "memory");
}
__device__ __forceinline__ void async_b32(unsigned lds, const void* g) {
  asm volatile("global_load_async_to_lds_b32 %0, %1, off" :: "v"(lds), "v"(g) : "memory");
}
__device__ __forceinline__ void wait_async() {
  asm volatile("s_wait_asynccnt 0x0" ::: "memory");
}
// CDNA5 Tensor Data Mover: 2D tile load, 4B elements, tile 256(x) x 64(y),
// tensor row stride 768 elements, LDS padded by 2 dwords every 256 dwords (-> stride 258)
__device__ __forceinline__ void tdm_load_2d_256x64(unsigned lds, unsigned long long ga) {
  v4i g0;
  g0.x = 1;                                                   // count=1 valid descriptor
  g0.y = (int)lds;                                            // lds_addr (bytes)
  g0.z = (int)(unsigned)(ga & 0xFFFFFFFFull);                 // global_addr[31:0]
  g0.w = (int)(((unsigned)(ga >> 32) & 0x01FFFFFFu) | 0x80000000u); // addr[56:32] | type=2
  v8i g1;
  g1[0] = (2 << 16) | (1 << 20) | (7 << 22) | (1 << 25);      // 4B, pad_en, every 256dw, +2dw
  g1[1] = (int)(256u << 16);                                  // tensor_dim0 = 256
  g1[2] = (int)(64u << 16);                                   // tensor_dim1 = 64
  g1[3] = (int)(256u << 16);                                  // tile_dim0 = 256
  g1[4] = 64;                                                 // tile_dim1 = 64 (tile_dim2=0)
  g1[5] = 768;                                                // tensor_dim0_stride = 768
  g1[6] = 0;
  g1[7] = 0;
  asm volatile("tensor_load_to_lds %0, %1" :: "s"(g0), "s"(g1) : "memory");
}

// ---------------------------------------------------------------- fp32 -> bf16 convert
__global__ void __launch_bounds__(256) k_f32_to_bf16(const float* __restrict__ src,
                                                     ushort_t* __restrict__ dst, int n) {
  for (int i = blockIdx.x * 256 + threadIdx.x; i < n; i += gridDim.x * 256) dst[i] = f2bf(src[i]);
}

// ---------------------------------------------------------------- layernorm (one block per row), bf16 out
__global__ void __launch_bounds__(256) k_layernorm_bf16(const float* __restrict__ x,
                                                        const float* __restrict__ g,
                                                        const float* __restrict__ bta,
                                                        ushort_t* __restrict__ out) {
  __shared__ float s1[256], s2[256];
  const int row = blockIdx.x;
  const float* xr = x + (size_t)row * EDIM;
  float a0 = 0.f, a1 = 0.f, vals[4];
#pragma unroll
  for (int i = 0; i < 4; i++) { float v = xr[threadIdx.x + i*256]; vals[i] = v; a0 += v; a1 += v*v; }
  s1[threadIdx.x] = a0; s2[threadIdx.x] = a1; __syncthreads();
  for (int s = 128; s > 0; s >>= 1) {
    if (threadIdx.x < s) { s1[threadIdx.x] += s1[threadIdx.x+s]; s2[threadIdx.x] += s2[threadIdx.x+s]; }
    __syncthreads();
  }
  float mean = s1[0] * (1.0f/EDIM);
  float var  = s2[0] * (1.0f/EDIM) - mean*mean;
  float rs   = rsqrtf(var + 1e-5f);
  ushort_t* outr = out + (size_t)row * EDIM;
#pragma unroll
  for (int i = 0; i < 4; i++) { int e = threadIdx.x + i*256; outr[e] = f2bf((vals[i]-mean)*rs*g[e] + bta[e]); }
}

// ---------------------------------------------------------------- gates: beta/decay = sigmoid(xn.w + b)
__global__ void __launch_bounds__(256) k_gates(const ushort_t* __restrict__ xn,
                                               const float* __restrict__ wgw, const float* __restrict__ bgw,
                                               const float* __restrict__ wgf, const float* __restrict__ bgf,
                                               float* __restrict__ gates) {
  __shared__ float s1[256], s2[256];
  const int row = blockIdx.x;
  const ushort_t* xr = xn + (size_t)row * EDIM;
  float a0 = 0.f, a1 = 0.f;
#pragma unroll
  for (int i = 0; i < 4; i++) { int e = threadIdx.x + i*256; float v = bf2f(xr[e]); a0 += v*wgw[e]; a1 += v*wgf[e]; }
  s1[threadIdx.x] = a0; s2[threadIdx.x] = a1; __syncthreads();
  for (int s = 128; s > 0; s >>= 1) {
    if (threadIdx.x < s) { s1[threadIdx.x] += s1[threadIdx.x+s]; s2[threadIdx.x] += s2[threadIdx.x+s]; }
    __syncthreads();
  }
  if (threadIdx.x == 0) {
    gates[row*2 + 0] = sigmoidf_(s1[0] + bgw[0]);   // beta
    gates[row*2 + 1] = sigmoidf_(s2[0] + bgf[0]);   // decay
  }
}

// ---------------------------------------------------------------- bf16 WMMA GEMM: C[Mr,N] = A[Mr,K] * Bw[N,K]^T
// 128x128 block tile, BLK_K=32, 8 waves each doing 64x32 (4x2 wmma tiles),
// double-buffered via global_load_async_to_lds_b128 (ASYNCcnt).
#define GEMM_LD 40   // halves; padded LDS row stride
#define GEMM_BUF (128 * GEMM_LD)
__device__ __forceinline__ v16bf frag_a_bf16(const ushort_t* S, int row0, int lane) {
  // 16-bit A 16x32: lanes 0-15 hold K{0..7,16..23}, lanes 16-31 hold K{8..15,24..31}
  int lr = lane & 15, hi = (lane >> 4) * 8;
  const ushort_t* p = S + (row0 + lr) * GEMM_LD + hi;
  return to_v16bf(*(const Q4*)p, *(const Q4*)(p + 16));
}
__device__ __forceinline__ v16bf frag_b_bf16(const ushort_t* S, int col0, int lane) {
  // 16-bit B 32x16: lanes 0-15 col n K{0..7,16..23}, lanes 16-31 col n K{8..15,24..31}
  int ln = lane & 15, hi = (lane >> 4) * 8;
  const ushort_t* p = S + (col0 + ln) * GEMM_LD + hi;
  return to_v16bf(*(const Q4*)p, *(const Q4*)(p + 16));
}

__global__ void __launch_bounds__(256) k_wmma_gemm(
    const ushort_t* __restrict__ A, const ushort_t* __restrict__ Bw,
    const float* __restrict__ bias, const float* __restrict__ res,
    float* __restrict__ outf, ushort_t* __restrict__ outh,
    int N, int K, int mode) {
  __shared__ ushort_t As[2][GEMM_BUF];
  __shared__ ushort_t Bs[2][GEMM_BUF];
  const int tid  = threadIdx.x;
  const int lane = tid & 31, wid = tid >> 5;
  const int wm = wid >> 2, wn = wid & 3;
  const int rowBase = blockIdx.y * 128;
  const int colBase = blockIdx.x * 128;
  v8f acc[4][2];
#pragma unroll
  for (int i = 0; i < 4; i++)
#pragma unroll
    for (int j = 0; j < 2; j++) acc[i][j] = v8f{0,0,0,0,0,0,0,0};

  const int r = tid >> 1;           // 128 rows, 2 threads/row
  const int h = (tid & 1) * 16;     // half-row (16 halves = 32B)
  const ushort_t* pa = A  + (size_t)(rowBase + r) * K + h;
  const ushort_t* pb = Bw + (size_t)(colBase + r) * K + h;
  unsigned lA[2], lB[2];
  lA[0] = lds_addr32(&As[0][r*GEMM_LD + h]);
  lA[1] = lds_addr32(&As[1][r*GEMM_LD + h]);
  lB[0] = lds_addr32(&Bs[0][r*GEMM_LD + h]);
  lB[1] = lds_addr32(&Bs[1][r*GEMM_LD + h]);

  // prologue: stage tile 0 asynchronously
  async_b128(lA[0], pa); async_b128(lA[0] + 32, pa + 8);
  async_b128(lB[0], pb); async_b128(lB[0] + 32, pb + 8);
  pa += 32; pb += 32;
  wait_async();
  __syncthreads();

  const int nk = K >> 5;
  int cur = 0;
  for (int it = 0; it < nk; ++it) {
    if (it + 1 < nk) {   // issue next tile while computing current
      int nx = cur ^ 1;
      async_b128(lA[nx], pa); async_b128(lA[nx] + 32, pa + 8);
      async_b128(lB[nx], pb); async_b128(lB[nx] + 32, pb + 8);
      __builtin_prefetch(pa + 32, 0, 1);
      __builtin_prefetch(pb + 32, 0, 1);
      pa += 32; pb += 32;
    }
    const ushort_t* Ab = As[cur];
    const ushort_t* Bb = Bs[cur];
    v16bf af[4], bfr[2];
#pragma unroll
    for (int mi = 0; mi < 4; mi++) af[mi] = frag_a_bf16(Ab, wm*64 + mi*16, lane);
#pragma unroll
    for (int ni = 0; ni < 2; ni++) bfr[ni] = frag_b_bf16(Bb, wn*32 + ni*16, lane);
#pragma unroll
    for (int mi = 0; mi < 4; mi++)
#pragma unroll
      for (int ni = 0; ni < 2; ni++)
        acc[mi][ni] = __builtin_amdgcn_wmma_f32_16x16x32_bf16(
            false, af[mi], false, bfr[ni], (short)0, acc[mi][ni], false, false);
    if (it + 1 < nk) wait_async();
    __syncthreads();
    cur ^= 1;
  }
  // epilogue
#pragma unroll
  for (int mi = 0; mi < 4; mi++) {
#pragma unroll
    for (int ni = 0; ni < 2; ni++) {
      int col  = colBase + wn*32 + ni*16 + (lane & 15);
      int row0 = rowBase + wm*64 + mi*16 + (lane >> 4) * 8;
      float bv = bias ? bias[col] : 0.0f;
#pragma unroll
      for (int v = 0; v < 8; v++) {
        size_t idx = (size_t)(row0 + v) * N + col;
        float val = acc[mi][ni][v] + bv;
        if (mode == 1)      outf[idx] = val + res[idx];               // +bias +residual
        else if (mode == 2) outh[idx] = f2bf(0.5f*val*(1.0f + erff(val*0.70710678118f))); // exact gelu
        else                outf[idx] = val;                          // raw
      }
    }
  }
}

// ---------------------------------------------------------------- proj epilogue: bias + l2norm(k), l2norm(q), tanh(v), in place
__global__ void __launch_bounds__(256) k_proj_epilogue(float* __restrict__ kqv,
                                                       const float* __restrict__ bk,
                                                       const float* __restrict__ bq,
                                                       const float* __restrict__ bv) {
  __shared__ float s1[256], s2[256];
  const int row = blockIdx.x, t = threadIdx.x;
  float* rp = kqv + (size_t)row * 768;
  float xk = rp[t]       + bk[t];
  float xq = rp[t + 256] + bq[t];
  float xv = rp[t + 512] + bv[t];
  s1[t] = xk*xk; s2[t] = xq*xq; __syncthreads();
  for (int s = 128; s > 0; s >>= 1) {
    if (t < s) { s1[t] += s1[t+s]; s2[t] += s2[t+s]; }
    __syncthreads();
  }
  float rk = 1.0f / fmaxf(sqrtf(s1[0]), 1e-12f);
  float rq = 1.0f / fmaxf(sqrtf(s2[0]), 1e-12f);
  rp[t] = xk * rk; rp[t + 256] = xq * rq; rp[t + 512] = tanhf(xv);
}

// ---------------------------------------------------------------- chunked scan (fp32 WMMA 16x16x4, state in LDS)
#define C_CH 64
#define RSL  64
#define LDM  258
#define LDKQ 258
#define LDVT 66
#define LDWM 65
#define SCAN_LDS_FLOATS (RSL*LDM + 2*C_CH*LDKQ + RSL*LDVT + C_CH*LDWM + 72 + 4*64)

__device__ __forceinline__ v8f wmma4(v2f a, v2f b, v8f c) {
  return __builtin_amdgcn_wmma_f32_16x16x4_f32(false, a, false, b, (short)0, c, false, false);
}
// A-frag (also B-as-transposed from row-major storage): f32 A 16x4 layout
__device__ __forceinline__ v2f frag_rm_f32(const float* S, int ld, int row0, int k0, int lane) {
  int lr = lane & 15, hi = (lane >> 4) * 2;
  const float* p = S + (row0 + lr) * ld + k0 + hi;
  v2f r = {p[0], p[1]}; return r;
}
// B-frag for a row-major B[k][n]
__device__ __forceinline__ v2f frag_bkn_f32(const float* S, int ld, int k0, int n0, int lane) {
  int ln = lane & 15, hi = (lane >> 4) * 2;
  const float* p = S + (k0 + hi) * ld + n0 + ln;
  v2f r = {p[0], p[ld]}; return r;
}
__device__ __forceinline__ v8f loadC_lds(const float* S, int ld, int m0, int n0, int lane) {
  int ln = lane & 15, mo = m0 + (lane >> 4) * 8;
  const float* p = S + mo * ld + n0 + ln;
  v8f c;
#pragma unroll
  for (int v = 0; v < 8; v++) c[v] = p[v*ld];
  return c;
}
__device__ __forceinline__ void storeC_lds(float* S, int ld, int m0, int n0, int lane, v8f c) {
  int ln = lane & 15, mo = m0 + (lane >> 4) * 8;
  float* p = S + mo * ld + n0 + ln;
#pragma unroll
  for (int v = 0; v < 8; v++) p[v*ld] = c[v];
}

__global__ void __launch_bounds__(256) k_scan(const float* __restrict__ kqv,
                                              const float* __restrict__ gates,
                                              const float* __restrict__ mem_in,
                                              ushort_t* __restrict__ ro,
                                              float* __restrict__ mem_out) {
  extern __shared__ float smem[];
  float* Msl = smem;                   // [RSL][LDM]  fp32 memory slice (rows r0..r0+63)
  float* Kc  = Msl + RSL*LDM;          // [C_CH][LDKQ]
  float* Qc  = Kc  + C_CH*LDKQ;        // [C_CH][LDKQ]
  float* Vt  = Qc  + C_CH*LDKQ;        // [RSL][LDVT]  VslT[i][j] = v_j[r0+i]
  float* Wm  = Vt  + RSL*LDVT;         // [C_CH][LDWM] Kdot then masked/scaled W
  float* Pp  = Wm  + C_CH*LDWM;        // 72: prefix decay products P_0..P_64
  float* cj  = Pp + 72;                // 64: beta_j / P_{j+1}
  float* gj  = cj + 64;                // 64: P_C * cj
  float* dbf = gj + 64;                // 64: decay
  float* bbf = dbf + 64;               // 64: beta

  const int tid = threadIdx.x;
  const int lane = tid & 31, wid = tid >> 5;
  const int b  = blockIdx.x >> 2;
  const int r0 = (blockIdx.x & 3) * RSL;

  for (int idx = tid; idx < RSL*MDIM; idx += 256) {
    int i = idx >> 8, d = idx & 255;
    Msl[i*LDM + d] = mem_in[(size_t)b*MDIM*MDIM + (size_t)(r0+i)*MDIM + d];
  }
  __syncthreads();

  for (int ch = 0; ch < SS / C_CH; ch++) {
    const int base = b*SS + ch*C_CH;
    // K and Q chunk tiles via the Tensor Data Mover (one wave issues, TENSORcnt)
    if (wid == 0) {
      unsigned long long gaK = (unsigned long long)(const void*)(kqv + (size_t)base * 768);
      tdm_load_2d_256x64(lds_addr32(Kc), gaK);
      tdm_load_2d_256x64(lds_addr32(Qc), gaK + 1024ull);   // +256 floats -> q block
    }
    // V transpose-gather via per-lane async global->LDS b32
    for (int idx = tid; idx < RSL*C_CH; idx += 256) {
      int i = idx >> 6, j = idx & 63;
      async_b32(lds_addr32(&Vt[i*LDVT + j]),
                kqv + (size_t)(base + j) * 768 + 512 + r0 + i);
    }
    if (tid < C_CH) {
      bbf[tid] = gates[(base + tid)*2 + 0];
      dbf[tid] = gates[(base + tid)*2 + 1];
    }
    wait_async();
    __builtin_amdgcn_s_wait_tensorcnt(0);
    __syncthreads();
    if (tid == 0) {
      Pp[0] = 1.0f;
      for (int t = 0; t < C_CH; t++) Pp[t+1] = Pp[t] * dbf[t];
      float PC = Pp[C_CH];
      for (int j = 0; j < C_CH; j++) { float c = bbf[j] / Pp[j+1]; cj[j] = c; gj[j] = PC * c; }
    }
    __syncthreads();

    // Kdot[j][t] = k_j . q_t   (inner 256, fp32 wmma)
    for (int tile = wid; tile < 16; tile += 8) {
      int jt = (tile >> 2) * 16, tt = (tile & 3) * 16;
      v8f acc = v8f{0,0,0,0,0,0,0,0};
      for (int d0 = 0; d0 < MDIM; d0 += 4)
        acc = wmma4(frag_rm_f32(Kc, LDKQ, jt, d0, lane), frag_rm_f32(Qc, LDKQ, tt, d0, lane), acc);
      storeC_lds(Wm, LDWM, jt, tt, lane, acc);
    }
    __syncthreads();
    // W[j][t] = (j<t) * P_t * cj[j] * Kdot[j][t]
    for (int idx = tid; idx < C_CH*C_CH; idx += 256) {
      int j = idx >> 6, t = idx & 63;
      Wm[j*LDWM + t] = (j < t) ? Pp[t] * cj[j] * Wm[j*LDWM + t] : 0.0f;
    }
    __syncthreads();
    // readout[i][t] = P_t * (M_prev[i,:] . q_t) + sum_j Vt[i][j] * W[j][t]
    for (int tile = wid; tile < 16; tile += 8) {
      int it = (tile >> 2) * 16, tt = (tile & 3) * 16;
      v8f acc = v8f{0,0,0,0,0,0,0,0};
      for (int d0 = 0; d0 < MDIM; d0 += 4)
        acc = wmma4(frag_rm_f32(Msl, LDM, it, d0, lane), frag_rm_f32(Qc, LDKQ, tt, d0, lane), acc);
      float pt = Pp[tt + (lane & 15)];
      acc = acc * pt;
      for (int j0 = 0; j0 < C_CH; j0 += 4)
        acc = wmma4(frag_rm_f32(Vt, LDVT, it, j0, lane), frag_bkn_f32(Wm, LDWM, j0, tt, lane), acc);
      int t  = tt + (lane & 15);
      int i0 = r0 + it + (lane >> 4) * 8;
      ushort_t* orow = ro + (size_t)(base + t) * MDIM + i0;
#pragma unroll
      for (int v = 0; v < 8; v++) orow[v] = f2bf(acc[v]);
    }
    __syncthreads();
    // scale Kc rows by gj[j] = P_C * cj[j]
    for (int idx = tid; idx < C_CH*MDIM; idx += 256) {
      int j = idx >> 8, d = idx & 255;
      Kc[j*LDKQ + d] *= gj[j];
    }
    __syncthreads();
    // M_new = P_C * M_prev + Vt @ Kc'
    float PC = Pp[C_CH];
    for (int tile = wid; tile < 64; tile += 8) {
      int it = (tile >> 4) * 16, dt = (tile & 15) * 16;
      v8f acc = loadC_lds(Msl, LDM, it, dt, lane);
      acc = acc * PC;
      for (int j0 = 0; j0 < C_CH; j0 += 4)
        acc = wmma4(frag_rm_f32(Vt, LDVT, it, j0, lane), frag_bkn_f32(Kc, LDKQ, j0, dt, lane), acc);
      storeC_lds(Msl, LDM, it, dt, lane, acc);
    }
    __syncthreads();
  }

  for (int idx = tid; idx < RSL*MDIM; idx += 256) {
    int i = idx >> 8, d = idx & 255;
    mem_out[(size_t)b*MDIM*MDIM + (size_t)(r0+i)*MDIM + d] = Msl[i*LDM + d];
  }
}

// ---------------------------------------------------------------- host
extern "C" void kernel_launch(void* const* d_in, const int* in_sizes, int n_in,
                              void* d_out, int out_size, void* d_ws, size_t ws_size,
                              hipStream_t stream) {
  (void)in_sizes; (void)n_in; (void)out_size; (void)ws_size;
  const float* x      = (const float*)d_in[0];
  const float* mem_in = (const float*)d_in[1];
  const float* w_k   = (const float*)d_in[2];
  const float* b_k   = (const float*)d_in[3];
  const float* w_q   = (const float*)d_in[4];
  const float* b_q   = (const float*)d_in[5];
  const float* w_v   = (const float*)d_in[6];
  const float* b_v   = (const float*)d_in[7];
  const float* w_out = (const float*)d_in[8];
  const float* b_out = (const float*)d_in[9];
  const float* w_gw  = (const float*)d_in[10];
  const float* b_gw  = (const float*)d_in[11];
  const float* w_gf  = (const float*)d_in[12];
  const float* b_gf  = (const float*)d_in[13];
  const float* ln1_g = (const float*)d_in[14];
  const float* ln1_b = (const float*)d_in[15];
  const float* ln2_g = (const float*)d_in[16];
  const float* ln2_b = (const float*)d_in[17];
  const float* w_f1  = (const float*)d_in[18];
  const float* b_f1  = (const float*)d_in[19];
  const float* w_f2  = (const float*)d_in[20];
  const float* b_f2  = (const float*)d_in[21];

  char* ws = (char*)d_ws;
  // zone A (dead after the mixer GEMM) -----------------------------------
  ushort_t* xn     = (ushort_t*)(ws + 0);          // 16,777,216 B
  ushort_t* wkqv_h = (ushort_t*)(ws + 16777216);   //  1,572,864 B
  ushort_t* wout_h = (ushort_t*)(ws + 18350080);   //    524,288 B
  float*    kqv    = (float*)   (ws + 18874368);   // 25,165,824 B
  float*    gates  = (float*)   (ws + 44040192);   //     65,536 B
  ushort_t* ro     = (ushort_t*)(ws + 44105728);   //  4,194,304 B
  // g1 aliases zone A once it is dead ------------------------------------
  ushort_t* g1     = (ushort_t*)(ws + 0);          // 67,108,864 B
  // live past the alias zone ---------------------------------------------
  float*    x_mid  = (float*)   (ws + 67108864);   // 33,554,432 B
  ushort_t* h_bf   = (ushort_t*)(ws + 100663296);  // 16,777,216 B
  ushort_t* wf1_h  = (ushort_t*)(ws + 117440512);  //  8,388,608 B
  ushort_t* wf2_h  = (ushort_t*)(ws + 125829120);  //  8,388,608 B

  // weights -> bf16
  k_f32_to_bf16<<<1024, 256, 0, stream>>>(w_k,   wkqv_h + 0,      MDIM*EDIM);
  k_f32_to_bf16<<<1024, 256, 0, stream>>>(w_q,   wkqv_h + 262144, MDIM*EDIM);
  k_f32_to_bf16<<<1024, 256, 0, stream>>>(w_v,   wkqv_h + 524288, MDIM*EDIM);
  k_f32_to_bf16<<<1024, 256, 0, stream>>>(w_out, wout_h,          EDIM*MDIM);
  k_f32_to_bf16<<<4096, 256, 0, stream>>>(w_f1,  wf1_h,           FFDIM*EDIM);
  k_f32_to_bf16<<<4096, 256, 0, stream>>>(w_f2,  wf2_h,           EDIM*FFDIM);

  // LN1, gates, projections (+ l2norm/tanh epilogue)
  k_layernorm_bf16<<<ROWS_TOT, 256, 0, stream>>>(x, ln1_g, ln1_b, xn);
  k_gates<<<ROWS_TOT, 256, 0, stream>>>(xn, w_gw, b_gw, w_gf, b_gf, gates);
  k_wmma_gemm<<<dim3(6, 64), 256, 0, stream>>>(xn, wkqv_h, nullptr, nullptr, kqv, nullptr, 768, EDIM, 0);
  k_proj_epilogue<<<ROWS_TOT, 256, 0, stream>>>(kqv, b_k, b_q, b_v);

  // chunked decay scan (fp32 wmma + TDM chunk staging)
  float* out_x   = (float*)d_out;
  float* out_mem = out_x + (size_t)ROWS_TOT * EDIM;
  const int scan_lds = SCAN_LDS_FLOATS * 4;   // ~233 KB, fits 320 KB CDNA5 LDS
  hipFuncSetAttribute(reinterpret_cast<const void*>(k_scan),
                      hipFuncAttributeMaxDynamicSharedMemorySize, scan_lds);
  k_scan<<<16, 256, scan_lds, stream>>>(kqv, gates, mem_in, ro, out_mem);

  // mixer: x_mid = x + readouts @ w_out^T + b_out
  k_wmma_gemm<<<dim3(8, 64), 256, 0, stream>>>(ro, wout_h, b_out, x, x_mid, nullptr, EDIM, MDIM, 1);
  // LN2 -> h (bf16)
  k_layernorm_bf16<<<ROWS_TOT, 256, 0, stream>>>(x_mid, ln2_g, ln2_b, h_bf);
  // FFN1: g1 = gelu(h @ w_f1^T + b_f1)  (bf16 out)
  k_wmma_gemm<<<dim3(32, 64), 256, 0, stream>>>(h_bf, wf1_h, b_f1, nullptr, nullptr, g1, FFDIM, EDIM, 2);
  // FFN2: out = x_mid + g1 @ w_f2^T + b_f2
  k_wmma_gemm<<<dim3(8, 64), 256, 0, stream>>>(g1, wf2_h, b_f2, x_mid, out_x, nullptr, EDIM, FFDIM, 1);
}
